// QLSTM_65481071395253
// MI455X (gfx1250) — matrix operand order repaired and checked by reading, not compiled, not run on hardware
//
#include <hip/hip_runtime.h>
#include <hip/hip_bf16.h>

typedef __bf16 bf16;
typedef __bf16 v16bf __attribute__((ext_vector_type(16)));
typedef __bf16 v8bf  __attribute__((ext_vector_type(8)));
typedef float  v8f   __attribute__((ext_vector_type(8)));

#define T_STEPS 512
#define BATCH   64
#define DIM     1024
#define HID     1024
#define KTOT    2048          // D + H
#define NWG     64            // workgroups in persistent scan kernel
#define BSTRIDE 2056          // padded LDS row stride (halves): 2048+8 -> bank-conflict-free b128

// ---------------- prep kernels ----------------

// zero barrier counters and the h double-buffer (h_{-1} = 0)
__global__ void zero_ws_kernel(int* __restrict__ cnt, bf16* __restrict__ hbuf) {
    int i = blockIdx.x * blockDim.x + threadIdx.x;
    if (i < T_STEPS) cnt[i] = 0;
    if (i < 2 * BATCH * HID) hbuf[i] = (bf16)0.0f;
}

// x (f32) -> bf16, flat
__global__ void cvt_x_kernel(const float* __restrict__ x, bf16* __restrict__ xb) {
    size_t i = (size_t)blockIdx.x * blockDim.x + threadIdx.x;
    if (i < (size_t)T_STEPS * BATCH * DIM) xb[i] = (bf16)x[i];
}

// W[gate] is [KTOT, HID] row-major f32; produce WT[gate][n][k] bf16 (k contiguous)
__global__ void cvt_w_kernel(const float* __restrict__ Wf, const float* __restrict__ Wi,
                             const float* __restrict__ Wg, const float* __restrict__ Wo,
                             bf16* __restrict__ WT) {
    size_t i = (size_t)blockIdx.x * blockDim.x + threadIdx.x;
    if (i >= (size_t)4 * HID * KTOT) return;
    int gate = (int)(i >> 21);            // HID*KTOT = 2^21
    int rem  = (int)(i & ((1u << 21) - 1));
    int n    = rem >> 11;                 // / KTOT
    int k    = rem & (KTOT - 1);
    const float* W = (gate == 0) ? Wf : (gate == 1) ? Wi : (gate == 2) ? Wg : Wo;
    WT[i] = (bf16)W[(size_t)k * HID + n];
}

// ---------------- persistent LSTM scan ----------------
// grid = 64 WGs x 256 threads (8 waves). WG wgid owns hidden units [wgid*16, wgid*16+16).
// Weights for this WG (4 gates x 16 units x 2048 K bf16 = 256 KB) are staged in LDS ONCE,
// then reused for all 512 steps (zero steady-state weight traffic to L2).
// Split-K wave mapping: wave w -> mtile = w&3 (batch tile of 16), khalf = w>>2
//   khalf 0: K = 0..1023   (x_t, streamed from global xb)
//   khalf 1: K = 1024..2047 (h_{t-1}, from L2-hot double buffer)
// Each wave computes ALL 4 gates for its m-tile over its K-half:
//   per k-step: 1 A fragment -> 4 WMMAs (A loaded exactly once per step, zero duplication).
// The two K-half partial sums land in lds_gates[2][..] and are summed in the cell update.

__global__ __launch_bounds__(256, 1)
void lstm_scan_kernel(const bf16* __restrict__ xb, const bf16* __restrict__ WT,
                      bf16* __restrict__ hbuf, int* __restrict__ cnt,
                      const float* __restrict__ bfp, const float* __restrict__ bip,
                      const float* __restrict__ bgp, const float* __restrict__ bop,
                      float* __restrict__ out) {
    const int wgid  = blockIdx.x;          // 0..63
    const int tid   = threadIdx.x;         // 0..255
    const int wave  = tid >> 5;            // 0..7
    const int lane  = tid & 31;
    const int mtile = wave & 3;            // 0..3  (batch tile of 16)
    const int khalf = wave >> 2;           // 0..1  (K half)

    __shared__ bf16  ldsB[4 * 16 * BSTRIDE];   // 263,168 B: WG's weight slab, padded rows
    __shared__ float lds_gates[2][BATCH][68];  // 34,816 B: per-K-half partial gate sums
    __shared__ float lds_c[BATCH][16];         // 4,096 B
    __shared__ float lds_bias[64];             // 256 B   (total ~295 KB of 320 KB WGP LDS)

    if (tid < 64) {
        int j = tid >> 4, u = tid & 15;
        const float* bp = (j == 0) ? bfp : (j == 1) ? bip : (j == 2) ? bgp : bop;
        lds_bias[tid] = bp[wgid * 16 + u];
    }
    for (int e = tid; e < BATCH * 16; e += 256) lds_c[e >> 4][e & 15] = 0.0f;

    // ---- stage B slab into LDS (once): rows = gate*16+unit, k contiguous, padded stride ----
    for (int chunk = tid; chunk < (4 * 16 * KTOT) / 8; chunk += 256) {
        int row = chunk >> 8;              // 256 8-half chunks per row
        int kc  = (chunk & 255) * 8;
        int j = row >> 4, u = row & 15;
        const bf16* s = WT + ((size_t)(j * HID + wgid * 16 + u)) * KTOT + kc;
        *(v8bf*)(&ldsB[row * BSTRIDE + kc]) = *(const v8bf*)s;
    }
    __syncthreads();

    // WMMA fragment addressing (this code's contiguous k order, same for A and B):
    // A 16x32: lane L -> row M = L&15, K = (L>>4)*16 + {0..15} contiguous.
    // B 32x16: lane L -> col N = L&15, K = (L>>4)*16 + {0..15} contiguous.
    const int ml = lane & 15;
    const int kg = lane >> 4;              // 0/1
    const int rowA = mtile * 16 + ml;      // batch row for A loads
    const bf16* B0 = ldsB + (0 * 16 + ml) * BSTRIDE + khalf * DIM + kg * 16;
    const bf16* B1 = ldsB + (1 * 16 + ml) * BSTRIDE + khalf * DIM + kg * 16;
    const bf16* B2 = ldsB + (2 * 16 + ml) * BSTRIDE + khalf * DIM + kg * 16;
    const bf16* B3 = ldsB + (3 * 16 + ml) * BSTRIDE + khalf * DIM + kg * 16;

    for (int t = 0; t < T_STEPS; ++t) {
        const bf16* hcur  = hbuf + (size_t)(t & 1) * BATCH * HID;
        bf16*       hnext = hbuf + (size_t)((t + 1) & 1) * BATCH * HID;

        v8f acc0 = {};
        v8f acc1 = {};
        v8f acc2 = {};
        v8f acc3 = {};

        // A source for this wave's K-half (wave-uniform select, no divergence)
        const bf16* A = (khalf == 0)
            ? xb + ((size_t)t * BATCH + rowA) * DIM + kg * 16
            : hcur + (size_t)rowA * HID + kg * 16;

        #pragma unroll 8
        for (int kh = 0; kh < 32; ++kh) {
            v8bf alo = *(const v8bf*)(A + kh * 32);
            v8bf ahi = *(const v8bf*)(A + kh * 32 + 8);
            v16bf a = __builtin_shufflevector(alo, ahi, 0,1,2,3,4,5,6,7,8,9,10,11,12,13,14,15);
            v16bf b0 = *(const v16bf*)(B0 + kh * 32);
            v16bf b1 = *(const v16bf*)(B1 + kh * 32);
            v16bf b2 = *(const v16bf*)(B2 + kh * 32);
            v16bf b3 = *(const v16bf*)(B3 + kh * 32);
            acc0 = __builtin_amdgcn_wmma_f32_16x16x32_bf16(false, a, false, b0,
                                                           (short)0, acc0, false, false);
            acc1 = __builtin_amdgcn_wmma_f32_16x16x32_bf16(false, a, false, b1,
                                                           (short)0, acc1, false, false);
            acc2 = __builtin_amdgcn_wmma_f32_16x16x32_bf16(false, a, false, b2,
                                                           (short)0, acc2, false, false);
            acc3 = __builtin_amdgcn_wmma_f32_16x16x32_bf16(false, a, false, b3,
                                                           (short)0, acc3, false, false);
        }

        // ---- D tiles -> LDS (C/D layout: VGPR r, M = r + 8*(lane>>4), N = lane&15) ----
        {
            int r0 = mtile * 16 + 8 * kg;
            #pragma unroll
            for (int r = 0; r < 8; ++r) {
                lds_gates[khalf][r0 + r][ 0 + ml] = acc0[r];
                lds_gates[khalf][r0 + r][16 + ml] = acc1[r];
                lds_gates[khalf][r0 + r][32 + ml] = acc2[r];
                lds_gates[khalf][r0 + r][48 + ml] = acc3[r];
            }
        }
        __syncthreads();

        // ---- elementwise LSTM cell on this WG's 16 hidden units ----
        for (int e = tid; e < BATCH * 16; e += 256) {
            int b = e >> 4, u = e & 15;
            float F = lds_gates[0][b][u]      + lds_gates[1][b][u]      + lds_bias[u];
            float I = lds_gates[0][b][16 + u] + lds_gates[1][b][16 + u] + lds_bias[16 + u];
            float G = lds_gates[0][b][32 + u] + lds_gates[1][b][32 + u] + lds_bias[32 + u];
            float O = lds_gates[0][b][48 + u] + lds_gates[1][b][48 + u] + lds_bias[48 + u];
            float fs = 1.0f / (1.0f + __expf(-F));
            float is = 1.0f / (1.0f + __expf(-I));
            float gs = tanhf(G);
            float os = 1.0f / (1.0f + __expf(-O));
            float c  = fs * lds_c[b][u] + is * gs;
            lds_c[b][u] = c;
            float h  = os * tanhf(c);
            int n = wgid * 16 + u;
            out[(size_t)t * (BATCH * HID) + b * HID + n] = h;
            hnext[(size_t)b * HID + n] = (bf16)h;
            if (t == T_STEPS - 1) {
                out[(size_t)T_STEPS * (BATCH * HID) + b * HID + n] = h;               // hT
                out[(size_t)T_STEPS * (BATCH * HID) + BATCH * HID + b * HID + n] = c; // cT
            }
        }

        // ---- device-wide step barrier (counter per step, zeroed each launch) ----
        __threadfence();
        __syncthreads();
        if (tid == 0) {
            __hip_atomic_fetch_add(&cnt[t], 1, __ATOMIC_RELEASE, __HIP_MEMORY_SCOPE_AGENT);
            while (__hip_atomic_load(&cnt[t], __ATOMIC_ACQUIRE, __HIP_MEMORY_SCOPE_AGENT) < NWG) {
                __builtin_amdgcn_s_sleep(2);
            }
        }
        __syncthreads();
        __threadfence();
    }
}

// ---------------- launcher ----------------

extern "C" void kernel_launch(void* const* d_in, const int* in_sizes, int n_in,
                              void* d_out, int out_size, void* d_ws, size_t ws_size,
                              hipStream_t stream) {
    const float* x   = (const float*)d_in[0];
    const float* Wf  = (const float*)d_in[1];
    const float* bfp = (const float*)d_in[2];
    const float* Wi  = (const float*)d_in[3];
    const float* bip = (const float*)d_in[4];
    const float* Wg  = (const float*)d_in[5];
    const float* bgp = (const float*)d_in[6];
    const float* Wo  = (const float*)d_in[7];
    const float* bop = (const float*)d_in[8];
    float* out = (float*)d_out;

    char* ws = (char*)d_ws;
    const size_t WT_BYTES = (size_t)4 * HID * KTOT * sizeof(bf16);           // 16 MB
    const size_t XB_BYTES = (size_t)T_STEPS * BATCH * DIM * sizeof(bf16);    // 64 MB
    const size_t HB_BYTES = (size_t)2 * BATCH * HID * sizeof(bf16);          // 256 KB
    bf16* WT   = (bf16*)ws;
    bf16* xb   = (bf16*)(ws + WT_BYTES);
    bf16* hbuf = (bf16*)(ws + WT_BYTES + XB_BYTES);
    int*  cnt  = (int*)(ws + WT_BYTES + XB_BYTES + HB_BYTES);

    // 1) zero counters + h0 (re-done every call: deterministic, graph-safe)
    {
        int n = 2 * BATCH * HID;  // covers cnt (512) too
        zero_ws_kernel<<<(n + 255) / 256, 256, 0, stream>>>(cnt, hbuf);
    }
    // 2) x -> bf16
    {
        size_t n = (size_t)T_STEPS * BATCH * DIM;
        cvt_x_kernel<<<(unsigned)((n + 255) / 256), 256, 0, stream>>>(x, xb);
    }
    // 3) W -> transposed bf16 WT[gate][n][k]
    {
        size_t n = (size_t)4 * HID * KTOT;
        cvt_w_kernel<<<(unsigned)((n + 255) / 256), 256, 0, stream>>>(Wf, Wi, Wg, Wo, WT);
    }
    // 4) persistent scan: 64 WGs x 256 threads, device barrier per step
    lstm_scan_kernel<<<NWG, 256, 0, stream>>>(xb, WT, hbuf, cnt,
                                              bfp, bip, bgp, bop, out);
}
